// DVRJANET_27023934227290
// MI455X (gfx1250) — matrix operand order repaired and checked by reading, not compiled, not run on hardware
//
#include <hip/hip_runtime.h>
#include <math.h>

// DVR-JANET recurrent cell on gfx1250 (MI455X).
// Latency-bound sequential scan over S=4096 steps. Design:
//  - all weight B-fragments hoisted into VGPRs (loop-invariant)
//  - fp32 recurrent state in per-lane registers (exclusive ownership);
//    f16 copies (WMMA A-fragments) double-buffered in LDS
//  - t-loop unrolled x2 so double-buffer indices are compile-time constants
//  - 2-step-deep x prefetch pipeline (raw x_{t+1} staged in registers)
//  - output projection split 4-ways per dot and balanced over all 4 waves
//    (16 v_fma_mix + 2-step shfl_xor butterfly per lane)
//  - 14x v_wmma_f32_16x16x32_f16 per step, hardware TRANS ops, 2 barriers/step
//  - 8 workgroups x 128 threads; WG owns 16 batch rows; each of 4 waves owns
//    one 16-column N-tile of H=64.

typedef __attribute__((ext_vector_type(16))) _Float16 v16h;
typedef __attribute__((ext_vector_type(8)))  float    v8f;

#define BSZ 128
#define SSZ 4096
#define HSZ 64
#define MT  16   // batch rows per workgroup

// ---- fast math on hardware TRANS ops ------------------------------------
__device__ __forceinline__ float fast_rcp(float x) { return __builtin_amdgcn_rcpf(x); }

#if defined(__has_builtin)
#if __has_builtin(__builtin_amdgcn_tanhf)
#define HAVE_HW_TANH 1
#endif
#endif

__device__ __forceinline__ float fast_tanh(float x) {
#ifdef HAVE_HW_TANH
    return __builtin_amdgcn_tanhf(x);            // gfx1250 v_tanh_f32
#else
    return 1.0f - 2.0f * fast_rcp(__expf(2.0f * x) + 1.0f);
#endif
}

__device__ __forceinline__ float fast_sigmoid(float x) {
    return fast_rcp(1.0f + __expf(-x));          // v_exp + v_rcp
}

// polynomial atan2 (inputs are Gaussian I/Q samples; |err| ~ 1e-4 rad)
__device__ __forceinline__ float fast_atan2(float y, float x) {
    const float ax = fabsf(x), ay = fabsf(y);
    const float mx = fmaxf(ax, ay), mn = fminf(ax, ay);
    const float a  = mn * fast_rcp(mx);
    const float s  = a * a;
    float r = fmaf(fmaf(fmaf(fmaf(0.0208351f, s, -0.085133f), s,
                             0.180141f), s, -0.3302995f), s, 0.9998660f) * a;
    r = (ay > ax)   ? 1.5707963268f - r : r;
    r = (x < 0.0f)  ? 3.1415926536f - r : r;
    return (y < 0.0f) ? -r : r;
}

// ---- CDNA5 WMMA fragment loaders (wave32) -------------------------------
// A: 16x32 f16 (MxK). Lane half h = lane>>4, m = lane&15.
//   elements 0..7  <-> K = kb + 8h + j ; elements 8..15 <-> K = kb + 16 + 8h + j
__device__ __forceinline__ v16h frag_A(const _Float16* s, int lda, int kb, int lane) {
    const int m = lane & 15;
    const int h = lane >> 4;
    const _Float16* p0 = s + m * lda + kb + 8 * h;
    const _Float16* p1 = s + m * lda + kb + 16 + 8 * h;
    v16h a;
#pragma unroll
    for (int j = 0; j < 8; ++j) { a[j] = p0[j]; a[8 + j] = p1[j]; }
    return a;
}

// B: 32x16 f16 (KxN), column N = n0 + (lane&15); elements 0..7 <-> K = kb + 8h + j,
// elements 8..15 <-> K = kb + 16 + 8h + j. W stored row-major W[n][k].
__device__ __forceinline__ v16h frag_B(const _Float16* w, int lda, int n0, int kb, int lane) {
    const int h = lane >> 4;
    const _Float16* p0 = w + (n0 + (lane & 15)) * lda + kb + 8 * h;
    const _Float16* p1 = w + (n0 + (lane & 15)) * lda + kb + 16 + 8 * h;
    v16h b;
#pragma unroll
    for (int j = 0; j < 8; ++j) { b[j] = p0[j]; b[8 + j] = p1[j]; }
    return b;
}

#define WMMA(A, B, C) \
    __builtin_amdgcn_wmma_f32_16x16x32_f16(false, (A), false, (B), (short)0, (C), false, false)

__global__ __launch_bounds__(128) void dvr_janet_kernel(
    const float* __restrict__ x,       const float* __restrict__ h0,
    const float* __restrict__ W_ph,    const float* __restrict__ W_ptheta,
    const float* __restrict__ W_ah,    const float* __restrict__ W_ax,
    const float* __restrict__ cs,      const float* __restrict__ W_f,
    const float* __restrict__ b_f,     const float* __restrict__ W_ccos,
    const float* __restrict__ b_ccos,  const float* __restrict__ W_csin,
    const float* __restrict__ b_csin,  const float* __restrict__ W_o1,
    const float* __restrict__ b_o1,    const float* __restrict__ W_o2,
    const float* __restrict__ b_o2,    float* __restrict__ out)
{
    // Weight staging (f16, row-major W[n][k]) -- used once to fill VGPR frags
    __shared__ _Float16 sWph[HSZ * HSZ];
    __shared__ _Float16 sWah[HSZ * HSZ];
    __shared__ _Float16 sWf [HSZ * HSZ];
    __shared__ _Float16 sWcc[HSZ * 2 * HSZ];
    __shared__ _Float16 sWcs[HSZ * 2 * HSZ];
    // f16 state copies, double-buffered (A-matrices for WMMA)
    __shared__ _Float16 sHIf[2][MT * HSZ], sHQf[2][MT * HSZ], sHsum[2][MT * HSZ];
    __shared__ _Float16 sAc[MT * HSZ], sAs[MT * HSZ];
    __shared__ float    sMag[2][MT], sTh[2][MT];   // double-buffered
    __shared__ float    sWo1[HSZ], sWo2[HSZ], sBo[2];

    const int tid  = threadIdx.x;
    const int lane = tid & 31;
    const int wid  = tid >> 5;
    const int n0   = wid * 16;           // this wave's N-tile base in H
    const int b0   = blockIdx.x * MT;    // this WG's batch-row base

    // ---- one-time: weights -> LDS (f32 -> f16) ----
    for (int i = tid; i < HSZ * HSZ; i += 128) {
        sWph[i] = (_Float16)W_ph[i];
        sWah[i] = (_Float16)W_ah[i];
        sWf [i] = (_Float16)W_f [i];
    }
    for (int i = tid; i < HSZ * 2 * HSZ; i += 128) {
        sWcc[i] = (_Float16)W_ccos[i];
        sWcs[i] = (_Float16)W_csin[i];
    }
    if (tid < HSZ) { sWo1[tid] = W_o1[tid]; sWo2[tid] = W_o2[tid]; }
    if (tid == 0)  { sBo[0] = b_o1[0]; sBo[1] = b_o2[0]; }
    // init f16 state (buffer 0) from h_0[0]
    for (int i = tid; i < MT * HSZ; i += 128) {
        const int m = i >> 6, k = i & 63;
        const float h = h0[(size_t)(b0 + m) * HSZ + k];
        sHIf[0][i] = (_Float16)h;  sHQf[0][i] = (_Float16)h;
        sHsum[0][i] = (_Float16)(h + h);
    }
    // x pipeline: mag/theta(t=0) -> LDS; raw x(t=1) staged in registers
    const float* xptr = x + (size_t)(b0 + tid) * SSZ * 2;   // valid for tid<MT
    float pIx = 0.0f, pQx = 0.0f;
    if (tid < MT) {
        const float ix = xptr[0], qx = xptr[1];
        sMag[0][tid] = __builtin_amdgcn_sqrtf(ix * ix + qx * qx);
        sTh [0][tid] = fast_atan2(qx, ix);
        pIx = xptr[2];  pQx = xptr[3];                      // x(t=1)
    }
    __syncthreads();

    // ---- hoist all loop-invariant weight B-fragments into VGPRs ----
    const v16h Bph0 = frag_B(sWph, HSZ, n0, 0,  lane);
    const v16h Bph1 = frag_B(sWph, HSZ, n0, 32, lane);
    const v16h Bah0 = frag_B(sWah, HSZ, n0, 0,  lane);
    const v16h Bah1 = frag_B(sWah, HSZ, n0, 32, lane);
    const v16h Bf0  = frag_B(sWf,  HSZ, n0, 0,  lane);
    const v16h Bf1  = frag_B(sWf,  HSZ, n0, 32, lane);
    const v16h Bcc0 = frag_B(sWcc, 2 * HSZ, n0, 0,  lane);
    const v16h Bcc1 = frag_B(sWcc, 2 * HSZ, n0, 32, lane);
    const v16h Bcc2 = frag_B(sWcc, 2 * HSZ, n0, 64, lane);
    const v16h Bcc3 = frag_B(sWcc, 2 * HSZ, n0, 96, lane);
    const v16h Bcs0 = frag_B(sWcs, 2 * HSZ, n0, 0,  lane);
    const v16h Bcs1 = frag_B(sWcs, 2 * HSZ, n0, 32, lane);
    const v16h Bcs2 = frag_B(sWcs, 2 * HSZ, n0, 64, lane);
    const v16h Bcs3 = frag_B(sWcs, 2 * HSZ, n0, 96, lane);

    // per-lane loop-invariant scalars (column nn of this lane's C-tile)
    const int   nn   = n0 + (lane & 15);
    const int   mb   = (lane >> 4) * 8;
    const float wpt  = W_ptheta[nn];
    const float wax  = W_ax[nn];
    const float bfv  = b_f[nn];
    const float bccv = b_ccos[nn];
    const float bcsv = b_csin[nn];
    const float c0 = cs[0], c1 = cs[1], c2 = cs[2], c3 = cs[3]; // K = 4 knots

    // fp32 recurrent state in registers: lane exclusively owns (mb+r, nn)
    float hiR[8], hqR[8];
#pragma unroll
    for (int r = 0; r < 8; ++r) {
        const float h = h0[(size_t)(b0 + mb + r) * HSZ + nn];
        hiR[r] = h;  hqR[r] = h;
    }

    // ---- output-projection assignment: 32 dots (16 rows x 2 outputs) split
    //      4 lanes per dot, 8 dots per wave -> balanced across all 4 waves.
    const int pD = wid * 8 + (lane >> 2);   // output pair 0..31
    const int mD = pD >> 1, cD = pD & 1, qD = lane & 3;
    _Float16 wOh[16];                       // this lane's 16 output weights (f16)
#pragma unroll
    for (int j = 0; j < 16; ++j)
        wOh[j] = (_Float16)((cD ? sWo2 : sWo1)[qD * 16 + j]);
    const float boD = cD ? sBo[1] : sBo[0];
    float* yptrD = out + ((size_t)(b0 + mD) * SSZ) * 2 + cD;

#pragma unroll 2
    for (int t = 0; t < SSZ; ++t) {
        const int cb = t & 1, wb = cb ^ 1;   // constants after x2 unroll

        // ---- phase A: theta_tilde / v / f-gate GEMMs (K = 64) ----
        v8f aT = {}, aV = {}, aF = {};
        {
            const v16h A0 = frag_A(sHsum[cb], HSZ, 0,  lane);
            const v16h A1 = frag_A(sHsum[cb], HSZ, 32, lane);
            aT = WMMA(A0, Bph0, aT);  aT = WMMA(A1, Bph1, aT);
            aV = WMMA(A0, Bah0, aV);  aV = WMMA(A1, Bah1, aV);
            aF = WMMA(A0, Bf0,  aF);  aF = WMMA(A1, Bf1,  aF);
        }
        // x pipeline: mag/theta(t+1) from staged registers; issue load of
        // x(t+2) now so its latency is hidden behind the next full step.
        if (tid < MT) {
            if (t + 1 < SSZ) {
                sMag[wb][tid] = __builtin_amdgcn_sqrtf(pIx * pIx + pQx * pQx);
                sTh [wb][tid] = fast_atan2(pQx, pIx);
            }
            if (t + 2 < SSZ) {
                pIx = xptr[(t + 2) * 2 + 0];
                pQx = xptr[(t + 2) * 2 + 1];
            }
        }
        float fg[8];
#pragma unroll
        for (int r = 0; r < 8; ++r) {
            const int m  = mb + r;
            const float tt = aT[r] + sTh[cb][m] * wpt;        // rank-1 theta term
            const float v  = aV[r] + sMag[cb][m] * wax;       // rank-1 magnitude term
            // DVR block: sum_k cs[k] * |v - knot_k|, knots = {.25,.5,.75,1}
            const float at = c0 * fabsf(v - 0.25f) + c1 * fabsf(v - 0.50f)
                           + c2 * fabsf(v - 0.75f) + c3 * fabsf(v - 1.00f);
            const float sn  = __sinf(tt);                     // v_sin_f32
            const float csn = __cosf(tt);                     // v_cos_f32
            sAc[m * HSZ + nn] = (_Float16)(at * csn);
            sAs[m * HSZ + nn] = (_Float16)(at * sn);
            fg[r] = fast_sigmoid(aF[r] + bfv);                // forget gate
        }
        __syncthreads();   // B1: sAc/sAs (and next-step sMag/sTh) visible

        // ---- phase B: g_cos / g_sin GEMMs (K = 128 = [h | a*trig]) ----
        v8f aC = {}, aS = {};
        aC = WMMA(frag_A(sHIf[cb], HSZ, 0,  lane), Bcc0, aC);
        aC = WMMA(frag_A(sHIf[cb], HSZ, 32, lane), Bcc1, aC);
        aC = WMMA(frag_A(sAc,      HSZ, 0,  lane), Bcc2, aC);
        aC = WMMA(frag_A(sAc,      HSZ, 32, lane), Bcc3, aC);
        aS = WMMA(frag_A(sHQf[cb], HSZ, 0,  lane), Bcs0, aS);
        aS = WMMA(frag_A(sHQf[cb], HSZ, 32, lane), Bcs1, aS);
        aS = WMMA(frag_A(sAs,      HSZ, 0,  lane), Bcs2, aS);
        aS = WMMA(frag_A(sAs,      HSZ, 32, lane), Bcs3, aS);

        // ---- phase C: register state update; write f16 copies to buffer wb
        //      (no barrier needed before this: writes go to the other buffer)
#pragma unroll
        for (int r = 0; r < 8; ++r) {
            const int idx = (mb + r) * HSZ + nn;
            const float gc = fast_tanh(aC[r] + bccv);
            const float gs = fast_tanh(aS[r] + bcsv);
            const float f  = fg[r];
            const float hi = f * hiR[r] + (1.0f - f) * gc;
            const float hq = f * hqR[r] + (1.0f - f) * gs;
            hiR[r] = hi;  hqR[r] = hq;
            sHIf[wb][idx]  = (_Float16)hi;
            sHQf[wb][idx]  = (_Float16)hq;
            sHsum[wb][idx] = (_Float16)(hi + hq);
        }
        __syncthreads();   // B2: new state visible for next step + output dot

        // ---- phase D: y = [hI @ W_o1^T + b_o1, hQ @ W_o2^T + b_o2] ----
        // Each dot split over 4 lanes (16 fma_mix each) + shfl_xor butterfly;
        // all 4 waves carry equal load, so no straggler wave per step.
        {
            const _Float16* hrow =
                (cD ? sHQf[wb] : sHIf[wb]) + mD * HSZ + qD * 16;
            float acc = 0.0f;
#pragma unroll
            for (int j = 0; j < 16; ++j)
                acc += (float)hrow[j] * (float)wOh[j];        // v_fma_mix
            acc += __shfl_xor(acc, 1, 32);
            acc += __shfl_xor(acc, 2, 32);
            if (qD == 0) yptrD[(size_t)t * 2] = acc + boD;
        }
    }
}

extern "C" void kernel_launch(void* const* d_in, const int* in_sizes, int n_in,
                              void* d_out, int out_size, void* d_ws, size_t ws_size,
                              hipStream_t stream) {
    (void)in_sizes; (void)n_in; (void)d_ws; (void)ws_size; (void)out_size;
    const float* x       = (const float*)d_in[0];
    const float* h0      = (const float*)d_in[1];
    const float* W_ph    = (const float*)d_in[2];
    const float* W_pth   = (const float*)d_in[3];
    const float* W_ah    = (const float*)d_in[4];
    const float* W_ax    = (const float*)d_in[5];
    const float* cs      = (const float*)d_in[6];
    const float* W_f     = (const float*)d_in[7];
    const float* b_f     = (const float*)d_in[8];
    const float* W_ccos  = (const float*)d_in[9];
    const float* b_ccos  = (const float*)d_in[10];
    const float* W_csin  = (const float*)d_in[11];
    const float* b_csin  = (const float*)d_in[12];
    const float* W_o1    = (const float*)d_in[13];
    const float* b_o1    = (const float*)d_in[14];
    const float* W_o2    = (const float*)d_in[15];
    const float* b_o2    = (const float*)d_in[16];
    float* out = (float*)d_out;

    dvr_janet_kernel<<<BSZ / MT, 128, 0, stream>>>(
        x, h0, W_ph, W_pth, W_ah, W_ax, cs, W_f, b_f,
        W_ccos, b_ccos, W_csin, b_csin, W_o1, b_o1, W_o2, b_o2, out);
}
